// SwitchTransformerLayer_18476949307738
// MI455X (gfx1250) — matrix operand order repaired and checked by reading, not compiled, not run on hardware
//
#include <hip/hip_runtime.h>

// ---------------------------------------------------------------------------
// Switch-Transformer layer for MI455X (gfx1250, wave32, WMMA).
// All heavy GEMMs use v_wmma_f32_16x16x32_bf16. MoE uses routed (top-1)
// dispatch — identical math to the reference's dense dispatch + one-hot mask.
// bf16 weight copies (w1+w2 = 128MB) fit in the 192MB global L2.
// Round 2: mask staged via LDS (coalesced b64 loads), GEMM K-slab = 64.
// ---------------------------------------------------------------------------

#define SS 2048
#define BB 2
#define DD 1024
#define HH 16
#define DKK 64
#define EE 8
#define FF 4096
#define TT (SS * BB)

typedef __attribute__((ext_vector_type(16))) __bf16 v16bf;
typedef __attribute__((ext_vector_type(8)))  __bf16 v8bf;
typedef __attribute__((ext_vector_type(8)))  float  v8f;

static __device__ __forceinline__ __bf16 f2bf(float f) {
  unsigned u = __builtin_bit_cast(unsigned, f);
  unsigned r = u + 0x7fffu + ((u >> 16) & 1u);
  unsigned short h = (unsigned short)(r >> 16);
  return __builtin_bit_cast(__bf16, h);
}
static __device__ __forceinline__ float bf2f(__bf16 b) {
  unsigned short h = __builtin_bit_cast(unsigned short, b);
  unsigned u = ((unsigned)h) << 16;
  return __builtin_bit_cast(float, u);
}
static __device__ __forceinline__ v16bf make16(const __bf16* p0, const __bf16* p1) {
  v8bf lo = *(const v8bf*)p0;
  v8bf hi = *(const v8bf*)p1;
  v16bf r;
#pragma unroll
  for (int i = 0; i < 8; ++i) { r[i] = lo[i]; r[i + 8] = hi[i]; }
  return r;
}

// ---------------------------------------------------------------------------
// fp32 -> bf16 conversion (weights), grid-stride
// ---------------------------------------------------------------------------
__global__ void cvt_bf16_kernel(const float* __restrict__ src,
                                __bf16* __restrict__ dst, size_t n) {
  size_t i = (size_t)blockIdx.x * blockDim.x + threadIdx.x;
  size_t stride = (size_t)gridDim.x * blockDim.x;
  for (; i < n; i += stride) dst[i] = f2bf(src[i]);
}

// ---------------------------------------------------------------------------
// LayerNorm over D=1024, one block per token, bf16 output
// ---------------------------------------------------------------------------
__global__ __launch_bounds__(256) void ln_bf16_kernel(
    const float* __restrict__ x, const float* __restrict__ g,
    const float* __restrict__ b, __bf16* __restrict__ z) {
  int t = blockIdx.x;
  const float* row = x + (size_t)t * DD;
  __shared__ float s1[256], s2[256];
  float lsum = 0.f, lsq = 0.f;
  for (int i = threadIdx.x; i < DD; i += 256) {
    float v = row[i];
    lsum += v; lsq += v * v;
  }
  s1[threadIdx.x] = lsum; s2[threadIdx.x] = lsq;
  __syncthreads();
  for (int o = 128; o > 0; o >>= 1) {
    if (threadIdx.x < o) {
      s1[threadIdx.x] += s1[threadIdx.x + o];
      s2[threadIdx.x] += s2[threadIdx.x + o];
    }
    __syncthreads();
  }
  float mu = s1[0] * (1.0f / DD);
  float var = s2[0] * (1.0f / DD) - mu * mu;
  float rstd = rsqrtf(var + 1e-5f);
  for (int i = threadIdx.x; i < DD; i += 256)
    z[(size_t)t * DD + i] = f2bf((row[i] - mu) * rstd * g[i] + b[i]);
}

// ---------------------------------------------------------------------------
// Generic WMMA bf16 GEMM: C = op(A[list?] @ B + bias), 128x64 block tile,
// 8 waves each own a 16x64 strip (4 accumulators). K staged in 64-wide slabs.
// ---------------------------------------------------------------------------
template <bool GATHER, bool RELU, bool RES, bool SCALE, bool BF16OUT>
__global__ __launch_bounds__(256) void wgemm_kernel(
    const __bf16* __restrict__ A, const __bf16* __restrict__ Bg,
    const float* __restrict__ bias, const float* __restrict__ resid,
    const float* __restrict__ scalev, const int* __restrict__ list,
    const int* __restrict__ cnt, int M, int N, int K,
    size_t bStrideE, size_t biasStrideE,
    float* __restrict__ outF, __bf16* __restrict__ outB) {
  int e = blockIdx.z;
  const __bf16* B = Bg + (size_t)e * bStrideE;
  const float* bb = bias + (size_t)e * biasStrideE;
  const int* lst = GATHER ? (list + (size_t)e * M) : nullptr;
  int Mact = GATHER ? cnt[e] : M;
  int m0 = blockIdx.y * 128;
  if (m0 >= Mact) return;
  int n0 = blockIdx.x * 64;

  __shared__ __bf16 As[128 * 64];  // 16KB
  __shared__ __bf16 Bs[64 * 64];   // 8KB, transposed: Bs[n][k]

  int tid = threadIdx.x;
  int w = tid >> 5, l = tid & 31;
  int lm = l & 15;
  int lh = (l >> 4) & 1;
  v8f acc[4] = {};

  // resolve this thread's A-tile source row once
  int ld_row = tid >> 1;
  int ld_seg = (tid & 1) * 32;
  int ld_mg = m0 + ld_row;
  int ld_rowA;
  if (GATHER) {
    int idx = (ld_mg < Mact) ? ld_mg : (Mact - 1);
    ld_rowA = lst[idx];
  } else {
    ld_rowA = ld_mg;
  }

  for (int kc = 0; kc < K; kc += 64) {
    __syncthreads();
    {  // A tile: 128 rows x 64 K, each thread one 32-elem segment
      const __bf16* src = A + (size_t)ld_rowA * K + kc + ld_seg;
      __bf16* dst = &As[ld_row * 64 + ld_seg];
#pragma unroll
      for (int j = 0; j < 4; ++j)
        *(v8bf*)(dst + j * 8) = *(const v8bf*)(src + j * 8);
    }
    {  // B tile transposed into Bs[n][k] (64 x 64)
      int n = tid & 63;
      int kg = tid >> 6;  // 0..3
#pragma unroll
      for (int i = 0; i < 16; ++i) {
        int kk = kg * 16 + i;
        Bs[n * 64 + kk] = B[(size_t)(kc + kk) * N + n0 + n];
      }
    }
    if (kc + 64 < K) __builtin_prefetch(&B[(size_t)(kc + 64) * N + n0], 0, 1);
    __syncthreads();

    int arow = w * 16 + lm;
    int a0 = lh ? 8 : 0;
    int b0 = lh ? 8 : 0;
#pragma unroll
    for (int c = 0; c < 2; ++c) {
      v16bf av = make16(&As[arow * 64 + c * 32 + a0],
                        &As[arow * 64 + c * 32 + a0 + 16]);
#pragma unroll
      for (int n = 0; n < 4; ++n) {
        int bcol = n * 16 + lm;
        v16bf bv = make16(&Bs[bcol * 64 + c * 32 + b0],
                          &Bs[bcol * 64 + c * 32 + b0 + 16]);
        acc[n] = __builtin_amdgcn_wmma_f32_16x16x32_bf16(
            false, av, false, bv, (short)0, acc[n], false, false);
      }
    }
  }

  // epilogue: C layout M = r + 8*lh, N = lm
#pragma unroll
  for (int n = 0; n < 4; ++n) {
    int col = n0 + n * 16 + lm;
    float bvv = bb[col];
#pragma unroll
    for (int r = 0; r < 8; ++r) {
      int mrow = m0 + w * 16 + r + lh * 8;
      bool valid = mrow < Mact;
      int orow;
      if (GATHER) {
        int idx = valid ? mrow : (Mact - 1);
        orow = lst[idx];
      } else {
        orow = mrow;
      }
      float vv = acc[n][r] + bvv;
      if (RELU) vv = fmaxf(vv, 0.f);
      if (SCALE) vv *= scalev[orow];
      if (RES) vv += resid[(size_t)orow * N + col];
      if (valid) {
        if (BF16OUT) outB[(size_t)orow * N + col] = f2bf(vv);
        else         outF[(size_t)orow * N + col] = vv;
      }
    }
  }
}

// ---------------------------------------------------------------------------
// Flash attention: grid (S/64, H, B), 128 threads (4 waves x 16 query rows).
// V transposed through LDS; mask tile staged through LDS with coalesced
// 8-byte loads; P transposed wave-locally through LDS (s_wait_dscnt).
// ---------------------------------------------------------------------------
__global__ __launch_bounds__(128) void attn_kernel(
    const __bf16* __restrict__ q, const __bf16* __restrict__ k,
    const __bf16* __restrict__ v, const unsigned char* __restrict__ mask,
    __bf16* __restrict__ o) {
  const float scale = 0.125f;  // 1/sqrt(64)
  int qt = blockIdx.x, h = blockIdx.y, b = blockIdx.z;
  int w = threadIdx.x >> 5, l = threadIdx.x & 31;
  int lm = l & 15, lh = (l >> 4) & 1;

  __shared__ __bf16 Vt[64 * 64];             // V transposed: [dk][key] (8KB)
  __shared__ __bf16 Pt[4][16 * 64];          // per-wave P tile (8KB)
  __shared__ unsigned long long Mt64[64 * 16];  // mask tile 64 x 128B (8KB)
  unsigned char* Mt = (unsigned char*)Mt64;

  int srow_frag = qt * 64 + w * 16 + lm;
  const __bf16* qrow = q + ((size_t)srow_frag * BB + b) * DD + h * 64;
  int a0 = lh ? 8 : 0;
  int b0 = lh ? 8 : 0;
  v16bf qa0 = make16(qrow + a0, qrow + a0 + 16);
  v16bf qa1 = make16(qrow + 32 + a0, qrow + 32 + a0 + 16);

  float mstat[8], lsum[8];
  v8f acco[4] = {};
#pragma unroll
  for (int r = 0; r < 8; ++r) { mstat[r] = -3.0e38f; lsum[r] = 0.f; }

  for (int kt = 0; kt < SS / 64; ++kt) {
    __syncthreads();
    {  // stage V tile transposed into LDS
      int key = threadIdx.x >> 1;
      int dkb = (threadIdx.x & 1) * 32;
      const __bf16* vrow =
          v + ((size_t)(kt * 64 + key) * BB + b) * DD + h * 64 + dkb;
#pragma unroll 4
      for (int j = 0; j < 32; ++j) Vt[(dkb + j) * 64 + key] = vrow[j];
    }
    {  // stage mask tile: rows qt*64.., cols kt*64.. for both b; 64x128 bytes
      int row = threadIdx.x >> 1;
      int half = (threadIdx.x & 1) * 8;  // in ull units: 0 or 8 (=64B)
      const unsigned long long* msrc = (const unsigned long long*)(
          mask + ((size_t)(qt * 64 + row) * SS + (size_t)kt * 64) * BB) + half;
#pragma unroll
      for (int j = 0; j < 8; ++j) Mt64[row * 16 + half + j] = msrc[j];
    }
    __syncthreads();

    v8f sc[4];
#pragma unroll
    for (int n = 0; n < 4; ++n) {
      int keyl = kt * 64 + n * 16 + lm;
      const __bf16* krow = k + ((size_t)keyl * BB + b) * DD + h * 64;
      v16bf kb0 = make16(krow + b0, krow + b0 + 16);
      v16bf kb1 = make16(krow + 32 + b0, krow + 32 + b0 + 16);
      v8f acc = {};
      acc = __builtin_amdgcn_wmma_f32_16x16x32_bf16(
          false, qa0, false, kb0, (short)0, acc, false, false);
      acc = __builtin_amdgcn_wmma_f32_16x16x32_bf16(
          false, qa1, false, kb1, (short)0, acc, false, false);
#pragma unroll
      for (int r = 0; r < 8; ++r) {
        int srl = w * 16 + r + lh * 8;       // local query row (0..63)
        int tcl = n * 16 + lm;               // local key col (0..63)
        float sv = acc[r] * scale;
        unsigned char mm = Mt[srl * 128 + tcl * 2 + b];
        sc[n][r] = mm ? sv : -3.0e38f;
      }
    }
    // online softmax over this 64-key slab
    float rmax[8], rsum[8];
#pragma unroll
    for (int r = 0; r < 8; ++r) {
      float m = fmaxf(fmaxf(sc[0][r], sc[1][r]), fmaxf(sc[2][r], sc[3][r]));
#pragma unroll
      for (int off = 1; off < 16; off <<= 1) m = fmaxf(m, __shfl_xor(m, off));
      rmax[r] = m;
    }
#pragma unroll
    for (int r = 0; r < 8; ++r) {
      float mnew = fmaxf(mstat[r], rmax[r]);
      float corr = __expf(mstat[r] - mnew);
      mstat[r] = mnew;
      lsum[r] *= corr;
#pragma unroll
      for (int d = 0; d < 4; ++d) acco[d][r] *= corr;
      rsum[r] = 0.f;
    }
#pragma unroll
    for (int n = 0; n < 4; ++n)
#pragma unroll
      for (int r = 0; r < 8; ++r) {
        float pv = __expf(sc[n][r] - mstat[r]);
        sc[n][r] = pv;
        rsum[r] += pv;
      }
#pragma unroll
    for (int r = 0; r < 8; ++r) {
      float s = rsum[r];
#pragma unroll
      for (int off = 1; off < 16; off <<= 1) s += __shfl_xor(s, off);
      lsum[r] += s;
    }
    // transpose P (C layout -> A layout) via wave-private LDS
    __bf16* P = &Pt[w][0];
#pragma unroll
    for (int n = 0; n < 4; ++n)
#pragma unroll
      for (int r = 0; r < 8; ++r)
        P[(r + lh * 8) * 64 + n * 16 + lm] = f2bf(sc[n][r]);
    asm volatile("s_wait_dscnt 0" ::: "memory");  // wave-local DS RAW
    const __bf16* Prow = P + lm * 64;
    v16bf pa0 = make16(Prow + a0, Prow + a0 + 16);
    v16bf pa1 = make16(Prow + 32 + a0, Prow + 32 + a0 + 16);
#pragma unroll
    for (int d = 0; d < 4; ++d) {
      const __bf16* Vrow = &Vt[(d * 16 + lm) * 64];
      v16bf vb0 = make16(Vrow + b0, Vrow + b0 + 16);
      v16bf vb1 = make16(Vrow + 32 + b0, Vrow + 32 + b0 + 16);
      acco[d] = __builtin_amdgcn_wmma_f32_16x16x32_bf16(
          false, pa0, false, vb0, (short)0, acco[d], false, false);
      acco[d] = __builtin_amdgcn_wmma_f32_16x16x32_bf16(
          false, pa1, false, vb1, (short)0, acco[d], false, false);
    }
  }
#pragma unroll
  for (int d = 0; d < 4; ++d)
#pragma unroll
    for (int r = 0; r < 8; ++r) {
      int srow = qt * 64 + w * 16 + r + lh * 8;
      int dk = d * 16 + lm;
      float vv = acco[d][r] / lsum[r];
      o[((size_t)srow * BB + b) * DD + h * 64 + dk] = f2bf(vv);
    }
}

// ---------------------------------------------------------------------------
// Router: one wave per token. logits = z2 @ w_gate + b_gate; softmax; top-1.
// ---------------------------------------------------------------------------
__global__ __launch_bounds__(32) void gate_kernel(
    const __bf16* __restrict__ z2, const float* __restrict__ wg,
    const float* __restrict__ bg, float* __restrict__ pmax_ws,
    float* __restrict__ tail, int* __restrict__ cnt_i, int* __restrict__ list) {
  int t = blockIdx.x, l = threadIdx.x;
  float acc[EE] = {};
  const __bf16* zr = z2 + (size_t)t * DD;
  for (int d = l; d < DD; d += 32) {
    float zv = bf2f(zr[d]);
    const float* wr = wg + (size_t)d * EE;
#pragma unroll
    for (int e = 0; e < EE; ++e) acc[e] += zv * wr[e];
  }
#pragma unroll
  for (int e = 0; e < EE; ++e)
#pragma unroll
    for (int off = 16; off > 0; off >>= 1) acc[e] += __shfl_xor(acc[e], off);
  if (l == 0) {
    float probs[EE];
    float mx = -3.0e38f;
#pragma unroll
    for (int e = 0; e < EE; ++e) {
      acc[e] += bg[e];
      mx = fmaxf(mx, acc[e]);
    }
    float sum = 0.f;
#pragma unroll
    for (int e = 0; e < EE; ++e) { probs[e] = __expf(acc[e] - mx); sum += probs[e]; }
    float inv = 1.f / sum;
    float pm = -1.f;
    int route = 0;
#pragma unroll
    for (int e = 0; e < EE; ++e) {
      probs[e] *= inv;
      if (probs[e] > pm) { pm = probs[e]; route = e; }
    }
    atomicAdd(&tail[route], 1.0f);                 // counts
#pragma unroll
    for (int e = 0; e < EE; ++e) atomicAdd(&tail[8 + e], probs[e]);  // rps
    tail[17 + t] = pm;                             // pmax output
    pmax_ws[t] = pm;
    int pos = atomicAdd(&cnt_i[route], 1);
    list[route * TT + pos] = t;
  }
}

__global__ void zero_kernel(float* __restrict__ tail, int* __restrict__ cnt_i) {
  int i = threadIdx.x;
  if (i < 17) tail[i] = 0.f;  // counts[8], rps[8], n_dropped (int 0 bits)
  if (i < EE) cnt_i[i] = 0;
}

// ---------------------------------------------------------------------------
extern "C" void kernel_launch(void* const* d_in, const int* in_sizes, int n_in,
                              void* d_out, int out_size, void* d_ws,
                              size_t ws_size, hipStream_t stream) {
  (void)in_sizes; (void)n_in; (void)out_size; (void)ws_size;
  const float* x    = (const float*)d_in[0];
  const float* ln1g = (const float*)d_in[1];
  const float* ln1b = (const float*)d_in[2];
  const float* ln2g = (const float*)d_in[3];
  const float* ln2b = (const float*)d_in[4];
  const float* wq   = (const float*)d_in[5];
  const float* bq   = (const float*)d_in[6];
  const float* wk   = (const float*)d_in[7];
  const float* bk   = (const float*)d_in[8];
  const float* wv   = (const float*)d_in[9];
  const float* bv   = (const float*)d_in[10];
  const float* wo   = (const float*)d_in[11];
  const float* bo   = (const float*)d_in[12];
  const float* wg   = (const float*)d_in[13];
  const float* bg   = (const float*)d_in[14];
  const float* w1   = (const float*)d_in[15];
  const float* b1   = (const float*)d_in[16];
  const float* w2   = (const float*)d_in[17];
  const float* b2   = (const float*)d_in[18];
  const unsigned char* mask = (const unsigned char*)d_in[19];

  float* outF = (float*)d_out;
  float* tail = outF + (size_t)TT * DD;

  char* p = (char*)d_ws;
  auto take = [&](size_t bytes) -> void* {
    void* r = (void*)p;
    p += (bytes + 255) & ~(size_t)255;
    return r;
  };
  __bf16* wqB = (__bf16*)take(sizeof(__bf16) * (size_t)DD * DD);
  __bf16* wkB = (__bf16*)take(sizeof(__bf16) * (size_t)DD * DD);
  __bf16* wvB = (__bf16*)take(sizeof(__bf16) * (size_t)DD * DD);
  __bf16* woB = (__bf16*)take(sizeof(__bf16) * (size_t)DD * DD);
  __bf16* w1B = (__bf16*)take(sizeof(__bf16) * (size_t)EE * DD * FF);
  __bf16* w2B = (__bf16*)take(sizeof(__bf16) * (size_t)EE * FF * DD);
  __bf16* z1B = (__bf16*)take(sizeof(__bf16) * (size_t)TT * DD);
  __bf16* qB  = (__bf16*)take(sizeof(__bf16) * (size_t)TT * DD);
  __bf16* kB  = (__bf16*)take(sizeof(__bf16) * (size_t)TT * DD);
  __bf16* vB  = (__bf16*)take(sizeof(__bf16) * (size_t)TT * DD);
  __bf16* oB  = (__bf16*)take(sizeof(__bf16) * (size_t)TT * DD);
  float*  x2  = (float*)take(sizeof(float) * (size_t)TT * DD);
  __bf16* z2B = (__bf16*)take(sizeof(__bf16) * (size_t)TT * DD);
  __bf16* hB  = (__bf16*)take(sizeof(__bf16) * (size_t)TT * FF);
  float*  pmaxW = (float*)take(sizeof(float) * (size_t)TT);
  int*    cntI  = (int*)take(sizeof(int) * EE);
  int*    listI = (int*)take(sizeof(int) * (size_t)EE * TT);

  zero_kernel<<<1, 32, 0, stream>>>(tail, cntI);

  cvt_bf16_kernel<<<1024, 256, 0, stream>>>(wq, wqB, (size_t)DD * DD);
  cvt_bf16_kernel<<<1024, 256, 0, stream>>>(wk, wkB, (size_t)DD * DD);
  cvt_bf16_kernel<<<1024, 256, 0, stream>>>(wv, wvB, (size_t)DD * DD);
  cvt_bf16_kernel<<<1024, 256, 0, stream>>>(wo, woB, (size_t)DD * DD);
  cvt_bf16_kernel<<<8192, 256, 0, stream>>>(w1, w1B, (size_t)EE * DD * FF);
  cvt_bf16_kernel<<<8192, 256, 0, stream>>>(w2, w2B, (size_t)EE * FF * DD);

  ln_bf16_kernel<<<TT, 256, 0, stream>>>(x, ln1g, ln1b, z1B);

  dim3 blk(256);
  dim3 gProj(DD / 64, TT / 128, 1);
  // q, k, v projections (bf16 out)
  wgemm_kernel<false, false, false, false, true><<<gProj, blk, 0, stream>>>(
      z1B, wqB, bq, nullptr, nullptr, nullptr, nullptr, TT, DD, DD, 0, 0,
      nullptr, qB);
  wgemm_kernel<false, false, false, false, true><<<gProj, blk, 0, stream>>>(
      z1B, wkB, bk, nullptr, nullptr, nullptr, nullptr, TT, DD, DD, 0, 0,
      nullptr, kB);
  wgemm_kernel<false, false, false, false, true><<<gProj, blk, 0, stream>>>(
      z1B, wvB, bv, nullptr, nullptr, nullptr, nullptr, TT, DD, DD, 0, 0,
      nullptr, vB);

  attn_kernel<<<dim3(SS / 64, HH, BB), 128, 0, stream>>>(qB, kB, vB, mask, oB);

  // x2 = x + o @ wo + bo   (fp32 out, residual)
  wgemm_kernel<false, false, true, false, false><<<gProj, blk, 0, stream>>>(
      oB, woB, bo, x, nullptr, nullptr, nullptr, TT, DD, DD, 0, 0, x2, nullptr);

  ln_bf16_kernel<<<TT, 256, 0, stream>>>(x2, ln2g, ln2b, z2B);

  gate_kernel<<<TT, 32, 0, stream>>>(z2B, wg, bg, pmaxW, tail, cntI, listI);

  // MoE expert GEMM 1: h[tok] = relu(z2[tok] @ w1[e] + b1[e])  (routed only)
  dim3 gM1(FF / 64, TT / 128, EE);
  wgemm_kernel<true, true, false, false, true><<<gM1, blk, 0, stream>>>(
      z2B, w1B, b1, nullptr, nullptr, listI, cntI, TT, FF, DD,
      (size_t)DD * FF, (size_t)FF, nullptr, hB);

  // MoE expert GEMM 2: out[tok] = x2[tok] + pmax[tok]*(h[tok] @ w2[e] + b2[e])
  dim3 gM2(DD / 64, TT / 128, EE);
  wgemm_kernel<true, false, true, true, false><<<gM2, blk, 0, stream>>>(
      hB, w2B, b2, x2, pmaxW, listI, cntI, TT, DD, FF,
      (size_t)FF * DD, (size_t)DD, outF, nullptr);
}
